// DeltaNetLayer_33844342293278
// MI455X (gfx1250) — compile-verified, matrix operand order checked by reading, not run on hardware
//
#include <hip/hip_runtime.h>
#include <hip/hip_bf16.h>
#include <math.h>

typedef __attribute__((ext_vector_type(16))) __bf16 v16bf;
typedef __attribute__((ext_vector_type(8)))  __bf16 v8bf;
typedef __attribute__((ext_vector_type(8)))  float  v8f;

#define B_DIM 8
#define T_DIM 2048
#define D_DIM 512
#define NTILES 4   // 16x64 output strip per wave

// ---------------------------------------------------------------------------
// bf16x3 split pre-pass: f = hi + lo (both bf16).  Bandwidth-bound, done once
// per operand so the GEMM inner loop carries no conversion VALU.
// ---------------------------------------------------------------------------
__global__ __launch_bounds__(256)
void split_bf16_kernel(const float* __restrict__ in, __bf16* __restrict__ hi,
                       __bf16* __restrict__ lo, size_t n) {
    size_t i = ((size_t)blockIdx.x * blockDim.x + threadIdx.x) * 4;
    const size_t stride = (size_t)gridDim.x * blockDim.x * 4;
    for (; i < n; i += stride) {
        float4 f = *(const float4*)(in + i);
        float fv[4] = {f.x, f.y, f.z, f.w};
#pragma unroll
        for (int j = 0; j < 4; ++j) {
            __bf16 h = (__bf16)fv[j];
            hi[i + j] = h;
            lo[i + j] = (__bf16)(fv[j] - (float)h);
        }
    }
}

// Load 16 bf16 in the WMMA 16-bit operand layout:
// halves 0..7 <- k0 + ksel*8 + {0..7}, halves 8..15 <- +16 further.
__device__ __forceinline__ v16bf ld16(const __bf16* p) {
    v8bf a = *(const v8bf*)(p);
    v8bf b = *(const v8bf*)(p + 16);
    return __builtin_shufflevector(a, b, 0, 1, 2, 3, 4, 5, 6, 7,
                                   8, 9, 10, 11, 12, 13, 14, 15);
}

// ---------------------------------------------------------------------------
// GEMM: Out[M,N] = A[M,K] * W[N,K]^T + bias[N], operands pre-split bf16 hi/lo.
// One wave per 16x64 strip: A regs feed 12 WMMAs per K=32 slab (bf16x3 x 4 N).
// ---------------------------------------------------------------------------
__global__ __launch_bounds__(256)
void gemm_wmma_bf16x3(const __bf16* __restrict__ Ah, const __bf16* __restrict__ Al,
                      const __bf16* __restrict__ Wh, const __bf16* __restrict__ Wl,
                      const float* __restrict__ bias, float* __restrict__ Out,
                      int M, int N, int K) {
    const int wave  = blockIdx.x * (blockDim.x >> 5) + (threadIdx.x >> 5);
    const int NtG   = N / (16 * NTILES);
    const int tileM = wave / NtG;
    const int ngrp  = wave % NtG;
    if (tileM * 16 >= M) return;

    const int lane = threadIdx.x & 31;
    const int r    = lane & 15;
    const int ksel = lane >> 4;

    const size_t aoff = (size_t)(tileM * 16 + r) * K + ksel * 8;
    const __bf16* arh = Ah + aoff;
    const __bf16* arl = Al + aoff;

    size_t woff[NTILES];
#pragma unroll
    for (int nt = 0; nt < NTILES; ++nt)
        woff[nt] = (size_t)((ngrp * NTILES + nt) * 16 + r) * K + ksel * 8;

    v8f acc[NTILES];
#pragma unroll
    for (int nt = 0; nt < NTILES; ++nt)
        acc[nt] = (v8f){0.f, 0.f, 0.f, 0.f, 0.f, 0.f, 0.f, 0.f};

    for (int k0 = 0; k0 < K; k0 += 32) {
        const v16bf ah = ld16(arh + k0);
        const v16bf al = ld16(arl + k0);
#pragma unroll
        for (int nt = 0; nt < NTILES; ++nt) {
            const v16bf bh = ld16(Wh + woff[nt] + k0);
            const v16bf bl = ld16(Wl + woff[nt] + k0);
            acc[nt] = __builtin_amdgcn_wmma_f32_16x16x32_bf16(
                false, ah, false, bh, (short)0, acc[nt], false, false);
            acc[nt] = __builtin_amdgcn_wmma_f32_16x16x32_bf16(
                false, ah, false, bl, (short)0, acc[nt], false, false);
            acc[nt] = __builtin_amdgcn_wmma_f32_16x16x32_bf16(
                false, al, false, bh, (short)0, acc[nt], false, false);
        }
    }

    // D layout: lanes 0-15 -> M=0..7 (vgpr 0..7), lanes 16-31 -> M=8..15; N=lane&15
    const int n  = lane & 15;
    const int mb = (lane >> 4) * 8;
#pragma unroll
    for (int nt = 0; nt < NTILES; ++nt) {
        const int ncol = (ngrp * NTILES + nt) * 16 + n;
        const float bv = bias ? bias[ncol] : 0.f;
        float* orow = Out + (size_t)(tileM * 16 + mb) * N + ncol;
#pragma unroll
        for (int v = 0; v < 8; ++v)
            orow[(size_t)v * N] = acc[nt][v] + bv;
    }
}

// ---------------------------------------------------------------------------
// Per-row: K <- K / max(||K||,eps) in place;  Beta = sigmoid(K . Wbeta + bbeta)
// ---------------------------------------------------------------------------
__global__ __launch_bounds__(256)
void norm_beta_kernel(float* __restrict__ Kbuf, const float* __restrict__ Wbeta,
                      const float* __restrict__ bbeta, float* __restrict__ Beta,
                      int R, int D) {
    const int row = blockIdx.x * (blockDim.x >> 5) + (threadIdx.x >> 5);
    if (row >= R) return;
    const int lane = threadIdx.x & 31;
    float* p = Kbuf + (size_t)row * D;

    float v[16];
    float ss = 0.f;
#pragma unroll
    for (int i = 0; i < 16; ++i) {
        v[i] = p[lane + i * 32];
        ss += v[i] * v[i];
    }
#pragma unroll
    for (int m = 16; m >= 1; m >>= 1) ss += __shfl_xor(ss, m, 32);

    const float scale = 1.f / fmaxf(sqrtf(ss), 1e-12f);

    float dot = 0.f;
#pragma unroll
    for (int i = 0; i < 16; ++i) {
        float kn = v[i] * scale;
        p[lane + i * 32] = kn;
        dot += kn * Wbeta[lane + i * 32];
    }
#pragma unroll
    for (int m = 16; m >= 1; m >>= 1) dot += __shfl_xor(dot, m, 32);

    if (lane == 0) Beta[row] = 1.f / (1.f + expf(-(dot + bbeta[0])));
}

// ---------------------------------------------------------------------------
// DeltaNet scan: rows of S are independent.  16 rows per WG, state in regs,
// k/q/v double-buffered in LDS, next step preloaded in regs + prefetched.
// ---------------------------------------------------------------------------
__device__ __forceinline__ float red16(float x) {
    x += __shfl_xor(x, 1, 32);
    x += __shfl_xor(x, 2, 32);
    x += __shfl_xor(x, 4, 32);
    x += __shfl_xor(x, 8, 32);
    return x;
}

__global__ __launch_bounds__(256)
void deltanet_scan_kernel(const float* __restrict__ Q, const float* __restrict__ K,
                          const float* __restrict__ V, const float* __restrict__ Beta,
                          float* __restrict__ O) {
    const int b    = blockIdx.y;
    const int rblk = blockIdx.x;
    const int tid  = threadIdx.x;
    const int lrow = tid >> 4;
    const int coff = (tid & 15) * 32;

    const float* Kp = K + (size_t)b * T_DIM * D_DIM;
    const float* Qp = Q + (size_t)b * T_DIM * D_DIM;
    const float* Vp = V + (size_t)b * T_DIM * D_DIM + rblk * 16;
    const float* Bp = Beta + (size_t)b * T_DIM;
    float* Op = O + (size_t)b * T_DIM * D_DIM + rblk * 16;

    __shared__ float kbuf[2][D_DIM];
    __shared__ float qbuf[2][D_DIM];
    __shared__ float vbuf[2][16];
    __shared__ float betab[2];

    float s[32];
#pragma unroll
    for (int j = 0; j < 32; ++j) s[j] = 0.f;

    float rk0 = Kp[tid], rk1 = Kp[256 + tid];
    float rq0 = Qp[tid], rq1 = Qp[256 + tid];
    float rv  = (tid < 16) ? Vp[tid] : 0.f;
    float rb  = (tid == 0) ? Bp[0] : 0.f;

    for (int t = 0; t < T_DIM; ++t) {
        const int buf = t & 1;
        kbuf[buf][tid] = rk0;  kbuf[buf][256 + tid] = rk1;
        qbuf[buf][tid] = rq0;  qbuf[buf][256 + tid] = rq1;
        if (tid < 16) vbuf[buf][tid] = rv;
        if (tid == 0) betab[buf] = rb;
        __syncthreads();

        const int tn = (t + 1 < T_DIM) ? t + 1 : t;
        const float* kn = Kp + (size_t)tn * D_DIM;
        const float* qn = Qp + (size_t)tn * D_DIM;
        rk0 = kn[tid]; rk1 = kn[256 + tid];
        rq0 = qn[tid]; rq1 = qn[256 + tid];
        if (tid < 16) rv = Vp[(size_t)tn * D_DIM + tid];
        if (tid == 0) rb = Bp[tn];
        const int tp = (t + 2 < T_DIM) ? t + 2 : t;
        __builtin_prefetch(Kp + (size_t)tp * D_DIM + tid, 0, 1);
        __builtin_prefetch(Qp + (size_t)tp * D_DIM + tid, 0, 1);

        const float beta = betab[buf];
        const float vr   = vbuf[buf][lrow];
        const float* kk  = &kbuf[buf][coff];
        const float* qq  = &qbuf[buf][coff];

        float skp = 0.f;
#pragma unroll
        for (int j = 0; j < 32; ++j) skp = fmaf(s[j], kk[j], skp);
        skp = red16(skp);

        const float coeff = beta * (vr - skp);

        float op = 0.f;
#pragma unroll
        for (int j = 0; j < 32; ++j) {
            s[j] = fmaf(coeff, kk[j], s[j]);
            op   = fmaf(s[j], qq[j], op);
        }
        op = red16(op);

        if ((tid & 15) == 0) Op[(size_t)t * D_DIM + lrow] = op;
    }
}

// ---------------------------------------------------------------------------
extern "C" void kernel_launch(void* const* d_in, const int* in_sizes, int n_in,
                              void* d_out, int out_size, void* d_ws, size_t ws_size,
                              hipStream_t stream) {
    const float* x     = (const float*)d_in[0];
    const float* Wq    = (const float*)d_in[1];
    const float* bq    = (const float*)d_in[2];
    const float* Wk    = (const float*)d_in[3];
    const float* bk    = (const float*)d_in[4];
    const float* Wv    = (const float*)d_in[5];
    const float* bv    = (const float*)d_in[6];
    const float* Wbeta = (const float*)d_in[7];
    const float* bbeta = (const float*)d_in[8];
    const float* Wo    = (const float*)d_in[9];
    const float* bo    = (const float*)d_in[10];
    float* out = (float*)d_out;

    const int M = B_DIM * T_DIM;       // 16384
    const int N = D_DIM;               // 512
    const int K = D_DIM;               // 512
    const size_t MN = (size_t)M * N;   // == M*K
    const size_t NK = (size_t)N * K;

    float* ws   = (float*)d_ws;
    float* Qr   = ws;                  // MN f32
    float* Kr   = ws + MN;             // MN f32 (normalized in place)
    float* Vr   = ws + 2 * MN;         // MN f32
    float* Os   = ws + 3 * MN;         // MN f32
    float* Beta = ws + 4 * MN;         // M  f32
    __bf16* xh  = (__bf16*)(ws + 4 * MN + M);   // MN bf16
    __bf16* xl  = xh + MN;                      // MN bf16
    __bf16* wh  = xl + MN;                      // NK bf16 (rewritten per GEMM)
    __bf16* wl  = wh + NK;                      // NK bf16
    __bf16* osh = (__bf16*)Qr;                  // alias: Q dead after scan
    __bf16* osl = osh + MN;

    const int tiles      = (M / 16) * (N / (16 * NTILES));
    const int gemmBlocks = tiles / 8;           // 8 waves per 256-thread block
    const int splitBig   = 4096;                // grid-stride covers MN
    const int splitSmall = 256;                 // covers NK

    split_bf16_kernel<<<splitBig, 256, 0, stream>>>(x, xh, xl, MN);

    split_bf16_kernel<<<splitSmall, 256, 0, stream>>>(Wq, wh, wl, NK);
    gemm_wmma_bf16x3<<<gemmBlocks, 256, 0, stream>>>(xh, xl, wh, wl, bq, Qr, M, N, K);

    split_bf16_kernel<<<splitSmall, 256, 0, stream>>>(Wk, wh, wl, NK);
    gemm_wmma_bf16x3<<<gemmBlocks, 256, 0, stream>>>(xh, xl, wh, wl, bk, Kr, M, N, K);

    split_bf16_kernel<<<splitSmall, 256, 0, stream>>>(Wv, wh, wl, NK);
    gemm_wmma_bf16x3<<<gemmBlocks, 256, 0, stream>>>(xh, xl, wh, wl, bv, Vr, M, N, K);

    norm_beta_kernel<<<M / 8, 256, 0, stream>>>(Kr, Wbeta, bbeta, Beta, M, D_DIM);

    dim3 sgrid(D_DIM / 16, B_DIM);
    deltanet_scan_kernel<<<sgrid, 256, 0, stream>>>(Qr, Kr, Vr, Beta, Os);

    split_bf16_kernel<<<splitBig, 256, 0, stream>>>(Os, osh, osl, MN);
    split_bf16_kernel<<<splitSmall, 256, 0, stream>>>(Wo, wh, wl, NK);
    gemm_wmma_bf16x3<<<gemmBlocks, 256, 0, stream>>>(osh, osl, wh, wl, bo, out, M, N, K);
}
